// GLAGCNModule_13700945674609
// MI455X (gfx1250) — matrix operand order repaired and checked by reading, not compiled, not run on hardware
//
#include <hip/hip_runtime.h>
#include <math.h>
#include <type_traits>

// ---------- problem constants ----------
#define NNODE 25
#define DIN   256
#define DHID  512
#define BT    4          // batches per workgroup
#define XS_STRIDE 264    // x rows: 256 cols + skew (528B, 16B aligned)
#define H_STRIDE  520    // h rows: 512 cols + skew (1040B, 16B aligned)
#define T_STRIDE  40     // 32 + skew (80B, 16B aligned) for na / staging
#define ACC_STRIDE 260   // floats

typedef _Float16 half8 __attribute__((ext_vector_type(8)));
typedef _Float16 v16h  __attribute__((ext_vector_type(16)));
typedef float    v8f   __attribute__((ext_vector_type(8)));

__constant__ int GIDX[53] = {
  0,1,2,3,4,5,6,7,8,9,10,11,12,13,14,15,16,17,18,19,20,21,22,23,24,
  0,1,2,3,4,5,6,7,8,
  0,9,10,11,12,13,14,
  6,7,8,12,13,14,
  3,4,5,9,10,11};
__constant__ int GOFF[5] = {0,25,34,41,47};
__constant__ int GN[5]   = {25,9,7,6,6};

// A-fragment (16-bit A 16x32): lane row = l&15; K = 16*hi+e
__device__ __forceinline__ v16h ld_afrag(const _Float16* rowp, int hi) {
  half8 lo = *(const half8*)(rowp + 16*hi);
  half8 hh = *(const half8*)(rowp + 16*hi + 8);
  return __builtin_shufflevector(lo, hh, 0,1,2,3,4,5,6,7,8,9,10,11,12,13,14,15);
}
// B-fragment: 16 contiguous halves per lane
__device__ __forceinline__ v16h ld_bfrag(const _Float16* p) {
  const half8* q = (const half8*)p;
  half8 lo = q[0];
  half8 hh = q[1];
  return __builtin_shufflevector(lo, hh, 0,1,2,3,4,5,6,7,8,9,10,11,12,13,14,15);
}
__device__ __forceinline__ v8f wmma16(v16h a, v16h b, v8f c) {
  return __builtin_amdgcn_wmma_f32_16x16x32_f16(false, a, false, b, (short)0, c, false, false);
}
__device__ __forceinline__ v8f v8f_zero() {
  v8f z = {0.f,0.f,0.f,0.f,0.f,0.f,0.f,0.f};
  return z;
}

// ---------- weight pre-swizzle: f32 -> f16 in exact WMMA B-fragment order ----------
__global__ void prep_weights(const float* __restrict__ W1, const float* __restrict__ W2,
                             _Float16* __restrict__ W1h, _Float16* __restrict__ W2h) {
  int idx = blockIdx.x * blockDim.x + threadIdx.x;
  const int N1 = 5*32*8*512;
  if (idx < N1) {
    int e  = idx & 15;
    int l  = (idx >> 4) & 31;
    int kb = (idx >> 9) & 7;
    int nt = (idx >> 12) & 31;
    int g  = idx >> 17;
    int K  = 32*kb + 16*(l >> 4) + e;
    int nc = nt*16 + (l & 15);
    W1h[idx] = (_Float16)W1[(g*DIN + K)*DHID + nc];
  } else {
    int j = idx - N1;
    if (j < 5*16*16*512) {
      int e  = j & 15;
      int l  = (j >> 4) & 31;
      int kb = (j >> 9) & 15;
      int nt = (j >> 13) & 15;
      int g  = j >> 17;
      int K  = 32*kb + 16*(l >> 4) + e;
      int nc = nt*16 + (l & 15);
      W2h[j] = (_Float16)W2[(g*DHID + K)*DIN + nc];
    }
  }
}

// ---------- fused GCN kernel: BT batches per workgroup ----------
__global__ void __launch_bounds__(256) glagcn_kernel(
    const float* __restrict__ x, const float* __restrict__ adj,
    const float* __restrict__ b1, const float* __restrict__ b2,
    const float* __restrict__ fusion_w,
    const _Float16* __restrict__ W1h, const _Float16* __restrict__ W2h,
    float* __restrict__ out)
{
  extern __shared__ float smem_f[];
  float* acc    = smem_f;                        // BT*25*ACC_STRIDE
  float* b1g    = acc + BT*25*ACC_STRIDE;        // 512
  float* b2g    = b1g + DHID;                    // 256
  float* dinv_s = b2g + DIN;                     // BT*32
  _Float16* xs    = (_Float16*)(dinv_s + BT*32); // BT*26*XS_STRIDE
  _Float16* hbuf  = xs   + BT*26*XS_STRIDE;      // BT*32*H_STRIDE
  _Float16* na    = hbuf + BT*32*H_STRIDE;       // BT*32*T_STRIDE
  _Float16* sbase = na   + BT*32*T_STRIDE;       // 8 waves * 16*T_STRIDE staging

  const int tid  = threadIdx.x;
  const int wid  = tid >> 5;
  const int lane = tid & 31;
  const int hi   = lane >> 4;
  const int lr   = lane & 15;
  const int b0   = blockIdx.x * BT;
  _Float16* sb = sbase + wid * 16 * T_STRIDE;    // per-wave transposed staging [col][row]

  // softmax(fusion_w)
  float nw[5];
  {
    float f0 = fusion_w[0], f1 = fusion_w[1], f2 = fusion_w[2], f3 = fusion_w[3], f4 = fusion_w[4];
    float m = fmaxf(fmaxf(fmaxf(f0, f1), fmaxf(f2, f3)), f4);
    float e0 = expf(f0-m), e1 = expf(f1-m), e2 = expf(f2-m), e3 = expf(f3-m), e4 = expf(f4-m);
    float s = e0+e1+e2+e3+e4;
    nw[0]=e0/s; nw[1]=e1/s; nw[2]=e2/s; nw[3]=e3/s; nw[4]=e4/s;
  }

  // load x (f32 -> f16) for BT batches, zero rows, zero accumulators
  for (int t2 = 0; t2 < BT; ++t2) {
    const float* xb = x + (size_t)(b0 + t2) * NNODE * DIN;
    _Float16* xst = xs + t2 * 26 * XS_STRIDE;
    for (int u = tid; u < NNODE * DIN; u += 256)
      xst[(u >> 8) * XS_STRIDE + (u & 255)] = (_Float16)xb[u];
    for (int u = tid; u < DIN; u += 256)
      xst[25 * XS_STRIDE + u] = (_Float16)0.f;
    float* at = acc + t2 * 25 * ACC_STRIDE;
    for (int u = tid; u < NNODE * DIN; u += 256)
      at[(u >> 8) * ACC_STRIDE + (u & 255)] = 0.f;
  }
  __syncthreads();

  auto group_body = [&](auto MTC, int g, int n, const int* gi) {
    constexpr int MTT = decltype(MTC)::value;

    // ---- setup: per-batch degree / D^-1/2 ; biases ----
    if (tid < BT * 32) {
      int t2 = tid >> 5, i = tid & 31;
      const float* adjb = adj + (size_t)(b0 + t2) * (NNODE * NNODE);
      float dv = 0.f;
      if (i < n) {
        float s = 1.f;
        const float* ar = adjb + gi[i] * NNODE;
        for (int j2 = 0; j2 < n; ++j2) s += ar[gi[j2]];
        dv = (s > 0.f) ? (1.f / sqrtf(s)) : 0.f;
      }
      dinv_s[tid] = dv;
    }
    for (int u = tid; u < DHID; u += 256) b1g[u] = b1[g * DHID + u];
    if (tid < DIN) b2g[tid] = b2[g * DIN + tid];
    __syncthreads();

    // ---- per-batch norm_adj (f16, zero-padded 32x32) ----
    for (int u = tid; u < BT * 1024; u += 256) {
      int t2 = u >> 10, r = (u >> 5) & 31, c2 = u & 31;
      float v = 0.f;
      if (r < n && c2 < n) {
        const float* adjb = adj + (size_t)(b0 + t2) * (NNODE * NNODE);
        float a = adjb[gi[r] * NNODE + gi[c2]] + ((r == c2) ? 1.f : 0.f);
        v = dinv_s[t2 * 32 + r] * a * dinv_s[t2 * 32 + c2];
      }
      na[(t2 * 32 + r) * T_STRIDE + c2] = (_Float16)v;
    }
    // per-wave staging pad rows must be zero when MT==1 (mix reads K=0..31)
    if (MTT == 1) {
      for (int q = lane; q < 256; q += 32)
        sb[(q >> 4) * T_STRIDE + 16 + (q & 15)] = (_Float16)0.f;
    }
    __syncthreads();

    // ---- Phase 1: support chunk GEMM + mix, per 16-col chunk of DHID ----
    // B-fragments (8 x v16h) stay resident across BT*MT row tiles;
    // A-fragments are software-pipelined (load kb+1 before wmma kb).
    for (int nt = wid; nt < 32; nt += 8) {
      const _Float16* bb = W1h + (size_t)((g * 32 + nt) * 8) * 512 + lane * 16;
      v16h bw[8];
      #pragma unroll
      for (int kb = 0; kb < 8; ++kb) bw[kb] = ld_bfrag(bb + kb * 512);
      const int colN = nt * 16 + lr;
      const float bias = b1g[colN];
      #pragma unroll
      for (int t2 = 0; t2 < BT; ++t2) {
        const _Float16* xst = xs + t2 * 26 * XS_STRIDE;
        const _Float16* arow[MTT];
        #pragma unroll
        for (int mt = 0; mt < MTT; ++mt) {
          int row = mt * 16 + lr;
          arow[mt] = xst + ((row < n) ? gi[row] : 25) * XS_STRIDE;
        }
        v8f C[MTT];
        v16h a_cur[MTT];
        #pragma unroll
        for (int mt = 0; mt < MTT; ++mt) {
          C[mt] = v8f_zero();
          a_cur[mt] = ld_afrag(arow[mt], hi);
        }
        #pragma unroll
        for (int kb = 0; kb < 8; ++kb) {
          v16h a_nxt[MTT];
          if (kb < 7) {
            #pragma unroll
            for (int mt = 0; mt < MTT; ++mt)
              a_nxt[mt] = ld_afrag(arow[mt] + (kb + 1) * 32, hi);
          }
          #pragma unroll
          for (int mt = 0; mt < MTT; ++mt)
            C[mt] = wmma16(a_cur[mt], bw[kb], C[mt]);
          if (kb < 7) {
            #pragma unroll
            for (int mt = 0; mt < MTT; ++mt) a_cur[mt] = a_nxt[mt];
          }
        }
        // stage support chunk transposed: sb[col][row]
        #pragma unroll
        for (int mt = 0; mt < MTT; ++mt) {
          _Float16* dp = sb + lr * T_STRIDE + mt * 16 + hi * 8;
          #pragma unroll
          for (int v2 = 0; v2 < 8; ++v2) dp[v2] = (_Float16)C[mt][v2];
        }
        // mix: h = relu(norm_adj @ support + b1)
        v16h bf = ld_bfrag(sb + lr * T_STRIDE + hi * 16);
        const _Float16* nat = na + t2 * 32 * T_STRIDE;
        _Float16* hbt = hbuf + t2 * 32 * H_STRIDE;
        #pragma unroll
        for (int mp = 0; mp < MTT; ++mp) {
          v16h a = ld_afrag(nat + (mp * 16 + lr) * T_STRIDE, hi);
          v8f cm = v8f_zero();
          cm = wmma16(a, bf, cm);
          #pragma unroll
          for (int v2 = 0; v2 < 8; ++v2) {
            int i = mp * 16 + hi * 8 + v2;
            float o = (i < n) ? fmaxf(cm[v2] + bias, 0.f) : 0.f;  // pad rows exactly zero
            hbt[i * H_STRIDE + colN] = (_Float16)o;
          }
        }
      }
    }
    __syncthreads();

    // ---- Phase 2: s2 chunk GEMM (K=512, two halves of 8 resident B-frags) + mix + fuse ----
    for (int nt = wid; nt < 16; nt += 8) {
      v8f C[BT][MTT];
      #pragma unroll
      for (int t2 = 0; t2 < BT; ++t2)
        #pragma unroll
        for (int mt = 0; mt < MTT; ++mt) C[t2][mt] = v8f_zero();
      #pragma unroll
      for (int kh = 0; kh < 2; ++kh) {
        const _Float16* bb = W2h + (size_t)((g * 16 + nt) * 16 + kh * 8) * 512 + lane * 16;
        v16h bw[8];
        #pragma unroll
        for (int kb = 0; kb < 8; ++kb) bw[kb] = ld_bfrag(bb + kb * 512);
        #pragma unroll
        for (int t2 = 0; t2 < BT; ++t2) {
          const _Float16* hbt = hbuf + t2 * 32 * H_STRIDE + kh * 256;
          #pragma unroll
          for (int mt = 0; mt < MTT; ++mt) {
            const _Float16* arow = hbt + (mt * 16 + lr) * H_STRIDE;
            v16h a_cur = ld_afrag(arow, hi);
            #pragma unroll
            for (int kb = 0; kb < 8; ++kb) {
              v16h a_nxt;
              if (kb < 7) a_nxt = ld_afrag(arow + (kb + 1) * 32, hi);
              C[t2][mt] = wmma16(a_cur, bw[kb], C[t2][mt]);
              if (kb < 7) a_cur = a_nxt;
            }
          }
        }
      }
      const int colN = nt * 16 + lr;
      const float bias = b2g[colN];
      const float w = nw[g];
      #pragma unroll
      for (int t2 = 0; t2 < BT; ++t2) {
        #pragma unroll
        for (int mt = 0; mt < MTT; ++mt) {
          _Float16* dp = sb + lr * T_STRIDE + mt * 16 + hi * 8;
          #pragma unroll
          for (int v2 = 0; v2 < 8; ++v2) dp[v2] = (_Float16)C[t2][mt][v2];
        }
        v16h bf = ld_bfrag(sb + lr * T_STRIDE + hi * 16);
        const _Float16* nat = na + t2 * 32 * T_STRIDE;
        float* at = acc + t2 * 25 * ACC_STRIDE;
        #pragma unroll
        for (int mp = 0; mp < MTT; ++mp) {
          v16h a = ld_afrag(nat + (mp * 16 + lr) * T_STRIDE, hi);
          v8f cm = v8f_zero();
          cm = wmma16(a, bf, cm);
          #pragma unroll
          for (int v2 = 0; v2 < 8; ++v2) {
            int i = mp * 16 + hi * 8 + v2;
            if (i < n)
              at[gi[i] * ACC_STRIDE + colN] += w * fmaxf(cm[v2] + bias, 0.f);
          }
        }
      }
    }
    __syncthreads();
  };

  for (int g = 0; g < 5; ++g) {
    const int n = GN[g];
    const int* gi = &GIDX[GOFF[g]];
    if (n > 16) group_body(std::integral_constant<int, 2>{}, g, n, gi);
    else        group_body(std::integral_constant<int, 1>{}, g, n, gi);
  }

  // write fused output (f32, coalesced)
  for (int t2 = 0; t2 < BT; ++t2) {
    float* ob = out + (size_t)(b0 + t2) * NNODE * DIN;
    const float* at = acc + t2 * 25 * ACC_STRIDE;
    for (int u = tid; u < NNODE * DIN; u += 256)
      ob[u] = at[(u >> 8) * ACC_STRIDE + (u & 255)];
  }
}

extern "C" void kernel_launch(void* const* d_in, const int* in_sizes, int n_in,
                              void* d_out, int out_size, void* d_ws, size_t ws_size,
                              hipStream_t stream) {
  const float* x   = (const float*)d_in[0];
  const float* adj = (const float*)d_in[1];
  const float* W1  = (const float*)d_in[2];
  const float* b1  = (const float*)d_in[3];
  const float* W2  = (const float*)d_in[4];
  const float* b2  = (const float*)d_in[5];
  const float* fw  = (const float*)d_in[6];
  float* out = (float*)d_out;

  _Float16* W1h = (_Float16*)d_ws;            // 5*32*8*512 halves
  _Float16* W2h = W1h + 5 * 32 * 8 * 512;     // 5*16*16*512 halves

  prep_weights<<<(2 * 655360) / 256, 256, 0, stream>>>(W1, W2, W1h, W2h);

  size_t lds_bytes =
      (size_t)(BT * 25 * ACC_STRIDE + DHID + DIN + BT * 32) * sizeof(float) +
      (size_t)(BT * 26 * XS_STRIDE + BT * 32 * H_STRIDE + BT * 32 * T_STRIDE + 8 * 16 * T_STRIDE) * sizeof(_Float16);

  glagcn_kernel<<<8192 / BT, 256, lds_bytes, stream>>>(x, adj, b1, b2, fw, W1h, W2h, out);
}
